// FullAttentionMultiHead_48593259987320
// MI455X (gfx1250) — matrix operand order repaired and checked by reading, not compiled, not run on hardware
//
#include <hip/hip_runtime.h>

#define D_MODEL 1024
#define NHEADS  16
#define HDIM    64
#define BATCH   4
#define SEQ     2048
#define MTOT    (BATCH * SEQ)   // 8192 rows

typedef __bf16 bf16;
typedef __attribute__((ext_vector_type(16))) __bf16 v16bf;
typedef __attribute__((ext_vector_type(8)))  __bf16 v8bf;
typedef __attribute__((ext_vector_type(4)))  __bf16 v4bf;
typedef __attribute__((ext_vector_type(8)))  float  v8f;

__device__ __forceinline__ v8f wmma_bf16(v16bf a, v16bf b, v8f c) {
  return __builtin_amdgcn_wmma_f32_16x16x32_bf16(false, a, false, b, (short)0, c,
                                                 false, false);
}

// A operand (16x32, row-major): lane L holds row L%16.
// K elements: {(L/16)*8 .. +7} and {16+(L/16)*8 .. +7}  (ISA 7.12.2)
__device__ __forceinline__ v16bf load_a_frag(const bf16* rowp, int lane) {
  const int hi = (lane >> 4) << 3;              // 0 or 8
  v8bf lo = *(const v8bf*)(rowp + hi);
  v8bf hh = *(const v8bf*)(rowp + hi + 16);
  v16bf r;
#pragma unroll
  for (int i = 0; i < 8; ++i) { r[i] = lo[i]; r[i + 8] = hh[i]; }
  return r;
}

// B operand (32x16): lane L holds column L%16.
// K elements: {(L/16)*8 .. +7} and {16+(L/16)*8 .. +7} (same as A)
__device__ __forceinline__ v16bf load_b_frag(const bf16* colp, int lane) {
  const int hi = (lane >> 4) << 3;              // 0 or 8
  v8bf lo = *(const v8bf*)(colp + hi);
  v8bf hh = *(const v8bf*)(colp + hi + 16);
  v16bf r;
#pragma unroll
  for (int i = 0; i < 8; ++i) { r[i] = lo[i]; r[i + 8] = hh[i]; }
  return r;
}

__global__ void cvt_f32_bf16(const float4* __restrict__ in, v4bf* __restrict__ out,
                             int n4) {
  int i = blockIdx.x * blockDim.x + threadIdx.x;
  int stride = gridDim.x * blockDim.x;
  for (; i < n4; i += stride) {
    float4 f = in[i];
    v4bf o;
    o[0] = (bf16)f.x; o[1] = (bf16)f.y; o[2] = (bf16)f.z; o[3] = (bf16)f.w;
    out[i] = o;
  }
}

// C[m][n] = sum_k A[m][k] * W[n][k] + bias[n];  M=8192, N=K=1024.
// Per-wave tile: 32 rows x 64 cols; ping-pong double buffering over k (no copies).
// mode 0: write bf16 Q [b][h][s][d]   1: bf16 K [b][h][s][d]
// mode 2: write bf16 Vt [b][h][d][s]  3: write f32 [m][n]
__global__ void __launch_bounds__(256)
gemm_xwT(const bf16* __restrict__ A, const bf16* __restrict__ W,
         const float* __restrict__ bias, void* __restrict__ out, int mode) {
  const int lane = threadIdx.x & 31;
  const int wv   = threadIdx.x >> 5;
  const int wid  = blockIdx.x * 8 + wv;
  const int mt   = wid >> 4;                    // 256 m-tiles of 32
  const int ng   = wid & 15;                    // 16 n-groups of 64
  const int m0   = mt << 5;
  const int n0   = ng << 6;
  const int l15  = lane & 15;

  const bf16* ap0 = A + (size_t)(m0 + l15) * D_MODEL;
  const bf16* ap1 = A + (size_t)(m0 + 16 + l15) * D_MODEL;
  const bf16* wp  = W + (size_t)(n0 + l15) * D_MODEL;

  v8f c[2][4];
#pragma unroll
  for (int mi = 0; mi < 2; ++mi)
#pragma unroll
    for (int t = 0; t < 4; ++t)
#pragma unroll
      for (int i = 0; i < 8; ++i) c[mi][t][i] = 0.f;

  v16bf a0A, a1A, bA[4], a0B, a1B, bB[4];

#define LOADSET(kkk, a0x, a1x, bx)                                              \
  do {                                                                          \
    a0x = load_a_frag(ap0 + (kkk), lane);                                       \
    a1x = load_a_frag(ap1 + (kkk), lane);                                       \
    _Pragma("unroll")                                                           \
    for (int t = 0; t < 4; ++t)                                                 \
      bx[t] = load_b_frag(wp + (size_t)t * 16 * D_MODEL + (kkk), lane);         \
  } while (0)

#define MMASET(a0x, a1x, bx)                                                    \
  do {                                                                          \
    _Pragma("unroll")                                                           \
    for (int t = 0; t < 4; ++t) {                                               \
      c[0][t] = wmma_bf16(a0x, bx[t], c[0][t]);                                 \
      c[1][t] = wmma_bf16(a1x, bx[t], c[1][t]);                                 \
    }                                                                           \
  } while (0)

  LOADSET(0, a0A, a1A, bA);
  for (int kk = 0; kk < D_MODEL - 64; kk += 64) {
    LOADSET(kk + 32, a0B, a1B, bB);
    MMASET(a0A, a1A, bA);
    LOADSET(kk + 64, a0A, a1A, bA);
    MMASET(a0B, a1B, bB);
  }
  LOADSET(D_MODEL - 32, a0B, a1B, bB);          // final pair, peeled
  MMASET(a0A, a1A, bA);
  MMASET(a0B, a1B, bB);
#undef LOADSET
#undef MMASET

  const int rbase = (lane >> 4) << 3;
#pragma unroll
  for (int mi = 0; mi < 2; ++mi) {
#pragma unroll
    for (int t = 0; t < 4; ++t) {
      const int n  = n0 + t * 16 + l15;
      const float bv = bias[n];
      const int h = n >> 6, d = n & 63;
      const int mrow = m0 + mi * 16 + rbase;    // first of 8 consecutive rows
      if (mode == 3) {
#pragma unroll
        for (int r = 0; r < 8; ++r)
          ((float*)out)[(size_t)(mrow + r) * D_MODEL + n] = c[mi][t][r] + bv;
      } else if (mode == 2) {
        // Vt[b][h][d][s]: 8 consecutive s for fixed d -> one 16B store
        const int bb = mrow >> 11, s = mrow & (SEQ - 1);
        v8bf pk;
#pragma unroll
        for (int r = 0; r < 8; ++r) pk[r] = (bf16)(c[mi][t][r] + bv);
        *(v8bf*)((bf16*)out +
                 (((size_t)(bb * NHEADS + h)) * HDIM + d) * SEQ + s) = pk;
      } else {
#pragma unroll
        for (int r = 0; r < 8; ++r) {
          const int m = mrow + r;
          const int bb = m >> 11, s = m & (SEQ - 1);
          ((bf16*)out)[(((size_t)(bb * NHEADS + h)) * SEQ + s) * HDIM + d] =
              (bf16)(c[mi][t][r] + bv);
        }
      }
    }
  }
}

// Flash attention: one wave owns a 16-row Q tile of one (b,h); KV chunk = 32.
// K fragments ping-pong double-buffered; V/mask loads issued under softmax math.
__global__ void __launch_bounds__(256)
flash_attn(const bf16* __restrict__ Q, const bf16* __restrict__ K,
           const bf16* __restrict__ Vt, const int* __restrict__ mask,
           bf16* __restrict__ AO) {
  __shared__ bf16 pstage[8 * 16 * 32];          // 1KB per wave
  const int lane = threadIdx.x & 31;
  const int wv   = threadIdx.x >> 5;
  const int wid  = blockIdx.x * 8 + wv;
  const int bh   = wid >> 7;                    // (b*16+h)
  const int qt   = wid & 127;
  const int b    = bh >> 4, h = bh & 15;
  const int q0   = qt << 4;
  const int l15  = lane & 15;
  const int rbase = (lane >> 4) << 3;
  bf16* pb = pstage + wv * 512;

  const bf16* qrow = Q + ((size_t)bh * SEQ + q0 + l15) * HDIM;
  v16bf aq0 = load_a_frag(qrow, lane);          // d 0..31
  v16bf aq1 = load_a_frag(qrow + 32, lane);     // d 32..63

  const bf16* kbase = K + (size_t)bh * SEQ * HDIM;
  const bf16* vbase = Vt + ((size_t)bh * HDIM + l15) * SEQ;

  float mi[8], li[8];
  v8f o[4];
#pragma unroll
  for (int r = 0; r < 8; ++r) { mi[r] = -1e30f; li[r] = 0.f; }
#pragma unroll
  for (int t = 0; t < 4; ++t)
#pragma unroll
    for (int i = 0; i < 8; ++i) o[t][i] = 0.f;

  auto load_k4 = [&](int kv, v16bf* bk) {
    const bf16* kp = kbase + (size_t)(kv + l15) * HDIM;
    bk[0] = load_b_frag(kp, lane);
    bk[1] = load_b_frag(kp + 32, lane);
    bk[2] = load_b_frag(kp + 16 * HDIM, lane);
    bk[3] = load_b_frag(kp + 16 * HDIM + 32, lane);
  };

  // One KV chunk: scores with bkC, prefetch next K into bkN, V/mask loads under
  // the softmax VALU work, then the P*V WMMAs.
  auto step = [&](int kv, v16bf* bkC, v16bf* bkN) {
    v8f c0, c1;
#pragma unroll
    for (int i = 0; i < 8; ++i) { c0[i] = 0.f; c1[i] = 0.f; }
    c0 = wmma_bf16(aq0, bkC[0], c0);
    c0 = wmma_bf16(aq1, bkC[1], c0);
    c1 = wmma_bf16(aq0, bkC[2], c1);
    c1 = wmma_bf16(aq1, bkC[3], c1);

    // prefetch next chunk's K (clamped to 0 on the last chunk; never consumed)
    load_k4(kv + 32 < SEQ ? kv + 32 : 0, bkN);

    v16bf bvf[4];
    const bf16* vp = vbase + kv;
#pragma unroll
    for (int t = 0; t < 4; ++t)
      bvf[t] = load_b_frag(vp + (size_t)t * 16 * SEQ, lane);
    const int mk0 = mask[b * SEQ + kv + l15];
    const int mk1 = mask[b * SEQ + kv + 16 + l15];

    float f0[8], f1[8], rm[8];
#pragma unroll
    for (int r = 0; r < 8; ++r) {
      f0[r] = mk0 ? c0[r] * 0.125f : -__builtin_inff();
      f1[r] = mk1 ? c1[r] * 0.125f : -__builtin_inff();
      rm[r] = fmaxf(f0[r], f1[r]);
    }
    for (int off = 1; off < 16; off <<= 1)
#pragma unroll
      for (int r = 0; r < 8; ++r) rm[r] = fmaxf(rm[r], __shfl_xor(rm[r], off));

    float p0[8], p1[8], rs[8];
#pragma unroll
    for (int r = 0; r < 8; ++r) {
      const float mn = fmaxf(mi[r], rm[r]);
      const float al = __expf(mi[r] - mn);
      p0[r] = __expf(f0[r] - mn);
      p1[r] = __expf(f1[r] - mn);
      rs[r] = p0[r] + p1[r];
      mi[r] = mn;
      li[r] *= al;
#pragma unroll
      for (int t = 0; t < 4; ++t) o[t][r] *= al;
    }
    for (int off = 1; off < 16; off <<= 1)
#pragma unroll
      for (int r = 0; r < 8; ++r) rs[r] += __shfl_xor(rs[r], off);
#pragma unroll
    for (int r = 0; r < 8; ++r) li[r] += rs[r];

    // Re-layout P (C-fragment) -> A-fragment via per-wave LDS tile.
#pragma unroll
    for (int r = 0; r < 8; ++r) {
      const int row = rbase + r;
      pb[row * 32 + l15]      = (bf16)p0[r];
      pb[row * 32 + 16 + l15] = (bf16)p1[r];
    }
    v16bf apf = load_a_frag(pb + l15 * 32, lane);

#pragma unroll
    for (int t = 0; t < 4; ++t) o[t] = wmma_bf16(apf, bvf[t], o[t]);
  };

  v16bf bkA[4], bkB[4];
  load_k4(0, bkA);
  for (int kv = 0; kv < SEQ; kv += 64) {        // 64 chunks -> 32 ping-pong pairs
    step(kv, bkA, bkB);
    step(kv + 32, bkB, bkA);
  }

#pragma unroll
  for (int r = 0; r < 8; ++r) {
    const float inv = li[r] > 0.f ? 1.f / li[r] : 0.f;
    const int s = q0 + rbase + r;
#pragma unroll
    for (int t = 0; t < 4; ++t)
      AO[((size_t)b * SEQ + s) * D_MODEL + h * HDIM + t * 16 + l15] =
          (bf16)(o[t][r] * inv);
  }
}

extern "C" void kernel_launch(void* const* d_in, const int* in_sizes, int n_in,
                              void* d_out, int out_size, void* d_ws, size_t ws_size,
                              hipStream_t stream) {
  (void)in_sizes; (void)n_in; (void)out_size; (void)ws_size;
  const float* x  = (const float*)d_in[0];
  const int* mask = (const int*)d_in[1];
  const float* Wq = (const float*)d_in[2];
  const float* bq = (const float*)d_in[3];
  const float* Wk = (const float*)d_in[4];
  const float* bk = (const float*)d_in[5];
  const float* Wv = (const float*)d_in[6];
  const float* bv = (const float*)d_in[7];
  const float* Wo = (const float*)d_in[8];
  const float* bo = (const float*)d_in[9];
  float* out = (float*)d_out;

  const size_t XN = (size_t)MTOT * D_MODEL;     // 8,388,608
  const size_t WN = (size_t)D_MODEL * D_MODEL;  // 1,048,576
  char* ws = (char*)d_ws;
  bf16* xb  = (bf16*)ws; ws += XN * 2;
  bf16* wqb = (bf16*)ws; ws += WN * 2;
  bf16* wkb = (bf16*)ws; ws += WN * 2;
  bf16* wvb = (bf16*)ws; ws += WN * 2;
  bf16* wob = (bf16*)ws; ws += WN * 2;
  bf16* Qb  = (bf16*)ws; ws += XN * 2;
  bf16* Kb  = (bf16*)ws; ws += XN * 2;
  bf16* Vt  = (bf16*)ws; ws += XN * 2;
  bf16* AO  = (bf16*)ws; ws += XN * 2;          // total ~92 MB

  cvt_f32_bf16<<<1024, 256, 0, stream>>>((const float4*)x,  (v4bf*)xb,  (int)(XN / 4));
  cvt_f32_bf16<<<256,  256, 0, stream>>>((const float4*)Wq, (v4bf*)wqb, (int)(WN / 4));
  cvt_f32_bf16<<<256,  256, 0, stream>>>((const float4*)Wk, (v4bf*)wkb, (int)(WN / 4));
  cvt_f32_bf16<<<256,  256, 0, stream>>>((const float4*)Wv, (v4bf*)wvb, (int)(WN / 4));
  cvt_f32_bf16<<<256,  256, 0, stream>>>((const float4*)Wo, (v4bf*)wob, (int)(WN / 4));

  // 256 m-tiles * 16 n-groups = 4096 waves / 8 per block = 512 blocks
  gemm_xwT<<<512, 256, 0, stream>>>(xb, wqb, bq, (void*)Qb, 0);
  gemm_xwT<<<512, 256, 0, stream>>>(xb, wkb, bk, (void*)Kb, 1);
  gemm_xwT<<<512, 256, 0, stream>>>(xb, wvb, bv, (void*)Vt, 2);

  // 4*16*128 q-tiles = 8192 waves / 8 = 1024 blocks
  flash_attn<<<1024, 256, 0, stream>>>(Qb, Kb, Vt, mask, AO);

  gemm_xwT<<<512, 256, 0, stream>>>(AO, wob, bo, (void*)out, 3);
}
